// ES_RNN_68358699483629
// MI455X (gfx1250) — matrix operand order repaired and checked by reading, not compiled, not run on hardware
//
#include <hip/hip_runtime.h>
#include <stdint.h>
#include <math.h>

// ---------------- problem constants (from reference) ----------------
#define T_LEN 1024
#define S_SEAS 168
#define H_DIM 128
#define WIN_L 168
#define OUT_L 24
#define N_WIN 833          // T - WIN - OUT + 1
#define WG_M 16            // windows per workgroup (one WMMA M-tile)
#define N_WGS 53           // ceil(833/16)
#define LDS_STRIDE 134     // 128 + 6 halfs pad -> 67-dword row stride (conflict-spreading)

typedef __attribute__((ext_vector_type(16))) _Float16 v16h;
typedef __attribute__((ext_vector_type(8)))  float    v8f;

// Branch-free activations: v_exp_f32 + raw v_rcp_f32 (no IEEE div fixup chain).
static __device__ __forceinline__ float fast_rcp(float v) { return __builtin_amdgcn_rcpf(v); }
static __device__ __forceinline__ float sigm_f(float v)  { return fast_rcp(1.0f + __expf(-v)); }
static __device__ __forceinline__ float tanh_f(float v)  { return fmaf(2.0f, fast_rcp(1.0f + __expf(-2.0f * v)), -1.0f); }

static __device__ __forceinline__ v8f vzero8() {
  v8f z = {0.f,0.f,0.f,0.f,0.f,0.f,0.f,0.f};
  return z;
}

static __device__ __forceinline__ v8f wmma_f16(v16h a, v16h b, v8f c) {
  // D = A(16x32 f16) * B(32x16 f16) + C(16x16 f32)
  return __builtin_amdgcn_wmma_f32_16x16x32_f16(false, a, false, b, (short)0, c, false, false);
}

// A-fragment (16x32 f16) from LDS row-major h[16][LDS_STRIDE]:
// lane L: row m = L%16, halfsel = L/16; VGPR v<4: k = kt*32 + 2v + hsel*8,
// v>=4: k = kt*32 + 16 + 2(v-4) + hsel*8 (per CDNA5 ISA 16-bit A table).
static __device__ __forceinline__ v16h load_afrag(const _Float16* sh, int m, int hsel, int kt) {
  v16h a;
#pragma unroll
  for (int v = 0; v < 8; ++v) {
    int base = (v < 4) ? (2 * v) : (16 + 2 * (v - 4));
    int k0 = kt * 32 + base + hsel * 8;
    a[2 * v]     = sh[m * LDS_STRIDE + k0];
    a[2 * v + 1] = sh[m * LDS_STRIDE + k0 + 1];
  }
  return a;
}

// B-fragment: pre-swizzled in global memory, 256 u32 per fragment, lane-major:
// u32 index = f*256 + lane*8 + v  -> two coalesced b128 loads per lane.
static __device__ __forceinline__ v16h load_bfrag(const uint32_t* __restrict__ fr, int f, int lane) {
  union { uint4 q[2]; v16h h; } u;
  const uint4* p = (const uint4*)(fr + (size_t)f * 256 + (size_t)lane * 8);
  u.q[0] = p[0];
  u.q[1] = p[1];
  return u.h;
}

// ---------------- kernel 1: sequential Holt-Winters scan ----------------
__global__ void es_scan_kernel(const float* __restrict__ x,
                               const float* __restrict__ lvl_raw,
                               const float* __restrict__ seas_raw,
                               const float* __restrict__ seas_params,
                               float* __restrict__ levels,
                               float* __restrict__ wsf) {
  if (threadIdx.x != 0 || blockIdx.x != 0) return;
  float a = sigm_f(lvl_raw[0]);
  float g = sigm_f(seas_raw[0]);
  for (int i = 0; i < S_SEAS; ++i) wsf[i] = __expf(seas_params[i]);
  wsf[S_SEAS] = wsf[0];
  float level = x[0] / wsf[0];
  levels[0] = level;
  for (int i = 1; i < T_LEN; ++i) {
    float wv = wsf[i];
    float nl = a * x[i] / wv + (1.0f - a) * level;
    float nw = g * x[i] / nl + (1.0f - g) * wv;
    if (S_SEAS + i < T_LEN) wsf[S_SEAS + i] = nw;
    levels[i] = nl;
    level = nl;
  }
}

// ---------------- kernel 2: labels + scalar loss ----------------
__global__ void labels_kernel(const float* __restrict__ x,
                              const float* __restrict__ levels,
                              const float* __restrict__ wsf,
                              const float* __restrict__ noise_lab,
                              float* __restrict__ out) {
  int idx = blockIdx.x * blockDim.x + threadIdx.x;
  if (idx == 0) out[2 * N_WIN * OUT_L] = 0.0f;  // level_var_loss
  if (idx < N_WIN * OUT_L) {
    int n = idx / OUT_L, o = idx % OUT_L;
    int t = n + WIN_L + o;
    float lev = levels[n + WIN_L];
    out[N_WIN * OUT_L + idx] = __logf(x[t] / (lev * wsf[t])) + noise_lab[idx];
  }
}

// ---------------- kernel 3: pack weights -> f16 B-fragment layout ----------------
// B(32x16 f16) matching A table order: lane -> n=lane%16, hsel=lane/16; VGPR v<4: K=kt*32+hsel*8+2v,
// v>=4: K=kt*32+16+hsel*8+2(v-4); each VGPR holds pair (K, K+1).
__global__ void pack_frags_kernel(const float* __restrict__ W, int nrows, int ntiles,
                                  uint32_t* __restrict__ dst) {
  int idx = blockIdx.x * blockDim.x + threadIdx.x;
  int total = ntiles * 4 * 256;
  if (idx >= total) return;
  int v    = idx & 7;
  int lane = (idx >> 3) & 31;
  int kt   = (idx >> 8) & 3;
  int nt   = idx >> 10;
  int n = nt * 16 + (lane & 15);
  int hsel = lane >> 4;
  int k0 = kt * 32 + ((v < 4) ? (2 * v) : (16 + 2 * (v - 4))) + hsel * 8;
  float w0 = 0.0f, w1 = 0.0f;
  if (n < nrows) { w0 = W[n * H_DIM + k0]; w1 = W[n * H_DIM + k0 + 1]; }
  union { _Float16 h[2]; uint32_t u; } pk;
  pk.h[0] = (_Float16)w0;
  pk.h[1] = (_Float16)w1;
  dst[idx] = pk.u;
}

// ---------------- kernel 4: fused 2-layer LSTM + head ----------------
// 53 WGs x 256 threads (8 wave32). WG owns 16 windows. Wave w owns gate/hidden
// columns [w*16, w*16+16): its 4 gate N-tiles {w, 8+w, 16+w, 24+w} are exactly
// i,f,g,o for those hidden units -> wave-local LSTM cell update, c in VGPRs.
//
// - All 48 weight B-fragments (384 VGPRs) register-resident across the loop.
// - h0/h1 double-buffered in LDS -> exactly ONE s_barrier per timestep.
// - A-fragments 2-deep software-pipelined so DS loads overlap WMMA groups.
__global__ __launch_bounds__(256) void lstm_head_kernel(
    const float* __restrict__ x, const float* __restrict__ noise_in,
    const float* __restrict__ levels, const float* __restrict__ wsf,
    const uint32_t* __restrict__ frWh0, const uint32_t* __restrict__ frWx1,
    const uint32_t* __restrict__ frWh1, const uint32_t* __restrict__ frTanh,
    const uint32_t* __restrict__ frLin,
    const float* __restrict__ Wx0, const float* __restrict__ b0,
    const float* __restrict__ b1, const float* __restrict__ tanh_b,
    const float* __restrict__ lin_b, float* __restrict__ out) {
  __shared__ _Float16 sh0[2][WG_M * LDS_STRIDE];   // layer0 hidden, double-buffered
  __shared__ _Float16 sh1[2][WG_M * LDS_STRIDE];   // layer1 hidden, double-buffered
  __shared__ float    ssq[WG_M * WIN_L];           // deseasonalized log inputs

  const int tid  = threadIdx.x;
  const int w    = tid >> 5;        // wave id 0..7
  const int lane = tid & 31;
  const int m16  = lane & 15;
  const int hs   = lane >> 4;
  const int wg   = blockIdx.x;
  const int colbase = w * 16 + m16; // this lane's hidden/gate column within its slice

  // ---- stage inputs: compute sq_in tile directly into LDS; zero h states ----
  for (int i = tid; i < WG_M * LDS_STRIDE; i += 256) {
    sh0[0][i] = (_Float16)0.0f; sh0[1][i] = (_Float16)0.0f;
    sh1[0][i] = (_Float16)0.0f; sh1[1][i] = (_Float16)0.0f;
  }
  for (int i = tid; i < WG_M * WIN_L; i += 256) {
    int m = i / WIN_L, t = i % WIN_L;
    int n = wg * WG_M + m; if (n > N_WIN - 1) n = N_WIN - 1;     // clamp padded rows
    ssq[i] = __logf(x[n + t] / (levels[n + WIN_L] * wsf[n + t])) + noise_in[n * WIN_L + t];
  }
  __syncthreads();

  // ---- per-lane time-invariant scalars ----
  float wx0s[4], b0s[4], b1s[4];
#pragma unroll
  for (int tau = 0; tau < 4; ++tau) {
    int cg = tau * 128 + colbase;   // global gate column (i,f,g,o blocks of 128)
    wx0s[tau] = Wx0[cg];
    b0s[tau]  = b0[cg];
    b1s[tau]  = b1[cg];
  }

  // ---- preload all 48 weight B-fragments into registers ----
  v16h BWh0[4][4], BWx1[4][4], BWh1[4][4];
#pragma unroll
  for (int tau = 0; tau < 4; ++tau) {
#pragma unroll
    for (int kt = 0; kt < 4; ++kt) {
      int f = (tau * 8 + w) * 4 + kt;
      BWh0[tau][kt] = load_bfrag(frWh0, f, lane);
      BWx1[tau][kt] = load_bfrag(frWx1, f, lane);
      BWh1[tau][kt] = load_bfrag(frWh1, f, lane);
    }
  }

  float c0[8], c1[8];
#pragma unroll
  for (int r = 0; r < 8; ++r) { c0[r] = 0.0f; c1[r] = 0.0f; }

  // ---- 168-step recurrence: ONE barrier per step ----
#pragma unroll 1
  for (int t = 0; t < WIN_L; ++t) {
    const int rb = t & 1;
    const _Float16* h0r = sh0[rb];
    _Float16*       h0w = sh0[rb ^ 1];
    const _Float16* h1r = sh1[rb];
    _Float16*       h1w = sh1[rb ^ 1];

    // ===== layer 0: gates0 = x_t * Wx0 + h0 @ Wh0^T + b0 =====
    v8f g0[4];
#pragma unroll
    for (int tau = 0; tau < 4; ++tau) g0[tau] = vzero8();
    {
      v16h aA = load_afrag(h0r, m16, hs, 0);
      v16h aB = load_afrag(h0r, m16, hs, 1);
#pragma unroll
      for (int tau = 0; tau < 4; ++tau) g0[tau] = wmma_f16(aA, BWh0[tau][0], g0[tau]);
      aA = load_afrag(h0r, m16, hs, 2);
#pragma unroll
      for (int tau = 0; tau < 4; ++tau) g0[tau] = wmma_f16(aB, BWh0[tau][1], g0[tau]);
      aB = load_afrag(h0r, m16, hs, 3);
#pragma unroll
      for (int tau = 0; tau < 4; ++tau) g0[tau] = wmma_f16(aA, BWh0[tau][2], g0[tau]);
#pragma unroll
      for (int tau = 0; tau < 4; ++tau) g0[tau] = wmma_f16(aB, BWh0[tau][3], g0[tau]);
    }

#pragma unroll
    for (int r = 0; r < 8; ++r) {
      int row = r + 8 * hs;
      float xt = ssq[row * WIN_L + t];
      float ig = g0[0][r] + xt * wx0s[0] + b0s[0];
      float fg = g0[1][r] + xt * wx0s[1] + b0s[1];
      float gg = g0[2][r] + xt * wx0s[2] + b0s[2];
      float og = g0[3][r] + xt * wx0s[3] + b0s[3];
      float c  = sigm_f(fg) * c0[r] + sigm_f(ig) * tanh_f(gg);
      c0[r] = c;
      h0w[row * LDS_STRIDE + colbase] = (_Float16)(sigm_f(og) * tanh_f(c));
    }
    __syncthreads();   // h0_new visible to all waves (sole barrier this step)

    // ===== layer 1: gates1 = h0_new @ Wx1^T + h1 @ Wh1^T + b1 =====
    v8f g1[4];
#pragma unroll
    for (int tau = 0; tau < 4; ++tau) g1[tau] = vzero8();
    {
      v16h aA = load_afrag(h0w, m16, hs, 0);
      v16h aB = load_afrag(h0w, m16, hs, 1);
#pragma unroll
      for (int tau = 0; tau < 4; ++tau) g1[tau] = wmma_f16(aA, BWx1[tau][0], g1[tau]);
      aA = load_afrag(h0w, m16, hs, 2);
#pragma unroll
      for (int tau = 0; tau < 4; ++tau) g1[tau] = wmma_f16(aB, BWx1[tau][1], g1[tau]);
      aB = load_afrag(h0w, m16, hs, 3);
#pragma unroll
      for (int tau = 0; tau < 4; ++tau) g1[tau] = wmma_f16(aA, BWx1[tau][2], g1[tau]);
      aA = load_afrag(h1r, m16, hs, 0);
#pragma unroll
      for (int tau = 0; tau < 4; ++tau) g1[tau] = wmma_f16(aB, BWx1[tau][3], g1[tau]);
      aB = load_afrag(h1r, m16, hs, 1);
#pragma unroll
      for (int tau = 0; tau < 4; ++tau) g1[tau] = wmma_f16(aA, BWh1[tau][0], g1[tau]);
      aA = load_afrag(h1r, m16, hs, 2);
#pragma unroll
      for (int tau = 0; tau < 4; ++tau) g1[tau] = wmma_f16(aB, BWh1[tau][1], g1[tau]);
      aB = load_afrag(h1r, m16, hs, 3);
#pragma unroll
      for (int tau = 0; tau < 4; ++tau) g1[tau] = wmma_f16(aA, BWh1[tau][2], g1[tau]);
#pragma unroll
      for (int tau = 0; tau < 4; ++tau) g1[tau] = wmma_f16(aB, BWh1[tau][3], g1[tau]);
    }

#pragma unroll
    for (int r = 0; r < 8; ++r) {
      float ig = g1[0][r] + b1s[0];
      float fg = g1[1][r] + b1s[1];
      float gg = g1[2][r] + b1s[2];
      float og = g1[3][r] + b1s[3];
      float c  = sigm_f(fg) * c1[r] + sigm_f(ig) * tanh_f(gg);
      c1[r] = c;
      h1w[(r + 8 * hs) * LDS_STRIDE + colbase] = (_Float16)(sigm_f(og) * tanh_f(c));
    }
    // h1_new writes are ordered for step t+1 by that step's barrier.
  }
  __syncthreads();

  // ---- head: feat = tanh(h1 @ tanh_W^T + tanh_b) ----
  // WIN_L is even -> final h1 lives in buffer 0; feat reuses sh0[0].
  {
    const _Float16* hf = sh1[WIN_L & 1];
    v8f fa = vzero8();
    v16h aA = load_afrag(hf, m16, hs, 0);
    v16h aB = load_afrag(hf, m16, hs, 1);
    fa = wmma_f16(aA, load_bfrag(frTanh, w * 4 + 0, lane), fa);
    aA = load_afrag(hf, m16, hs, 2);
    fa = wmma_f16(aB, load_bfrag(frTanh, w * 4 + 1, lane), fa);
    aB = load_afrag(hf, m16, hs, 3);
    fa = wmma_f16(aA, load_bfrag(frTanh, w * 4 + 2, lane), fa);
    fa = wmma_f16(aB, load_bfrag(frTanh, w * 4 + 3, lane), fa);
    float tb = tanh_b[colbase];
#pragma unroll
    for (int r = 0; r < 8; ++r)
      sh0[0][(r + 8 * hs) * LDS_STRIDE + colbase] = (_Float16)tanh_f(fa[r] + tb);
  }
  __syncthreads();

  // ---- head: out = feat @ lin_W^T + lin_b (waves 0,1 cover 32 >= 24 cols) ----
  if (w < 2) {
    v8f oa = vzero8();
#pragma unroll
    for (int kt = 0; kt < 4; ++kt) {
      v16h ao = load_afrag(sh0[0], m16, hs, kt);
      oa = wmma_f16(ao, load_bfrag(frLin, w * 4 + kt, lane), oa);
    }
    int o = w * 16 + m16;
    if (o < OUT_L) {
      float lb = lin_b[o];
#pragma unroll
      for (int r = 0; r < 8; ++r) {
        int n = wg * WG_M + r + 8 * hs;
        if (n < N_WIN) out[n * OUT_L + o] = oa[r] + lb;
      }
    }
  }
}

// ---------------- host-side launcher ----------------
extern "C" void kernel_launch(void* const* d_in, const int* in_sizes, int n_in,
                              void* d_out, int out_size, void* d_ws, size_t ws_size,
                              hipStream_t stream) {
  (void)in_sizes; (void)n_in; (void)out_size; (void)ws_size;
  const float* x        = (const float*)d_in[0];
  const float* lvl_raw  = (const float*)d_in[1];
  const float* seas_raw = (const float*)d_in[2];
  const float* seas_p   = (const float*)d_in[3];
  const float* noise_in = (const float*)d_in[4];
  const float* noise_la = (const float*)d_in[5];
  const float* Wx0      = (const float*)d_in[6];
  const float* Wh0      = (const float*)d_in[7];
  const float* b0       = (const float*)d_in[8];
  const float* Wx1      = (const float*)d_in[9];
  const float* Wh1      = (const float*)d_in[10];
  const float* b1       = (const float*)d_in[11];
  const float* tanhW    = (const float*)d_in[12];
  const float* tanhb    = (const float*)d_in[13];
  const float* linW     = (const float*)d_in[14];
  const float* linb     = (const float*)d_in[15];
  float* out = (float*)d_out;

  // workspace layout (bytes): levels 4KB | wsf 4KB | Wh0 128KB | Wx1 128KB |
  // Wh1 128KB | tanhW 32KB | linW 8KB  == ~432KB total
  char* ws = (char*)d_ws;
  float*    levels = (float*)(ws + 0);
  float*    wsf    = (float*)(ws + 4096);
  uint32_t* frWh0  = (uint32_t*)(ws + 8192);
  uint32_t* frWx1  = (uint32_t*)(ws + 8192 + 131072);
  uint32_t* frWh1  = (uint32_t*)(ws + 8192 + 2 * 131072);
  uint32_t* frTanh = (uint32_t*)(ws + 8192 + 3 * 131072);
  uint32_t* frLin  = (uint32_t*)(ws + 8192 + 3 * 131072 + 32768);

  es_scan_kernel<<<1, 32, 0, stream>>>(x, lvl_raw, seas_raw, seas_p, levels, wsf);

  pack_frags_kernel<<<(32 * 4 * 256) / 256, 256, 0, stream>>>(Wh0,  512, 32, frWh0);
  pack_frags_kernel<<<(32 * 4 * 256) / 256, 256, 0, stream>>>(Wx1,  512, 32, frWx1);
  pack_frags_kernel<<<(32 * 4 * 256) / 256, 256, 0, stream>>>(Wh1,  512, 32, frWh1);
  pack_frags_kernel<<<(8  * 4 * 256) / 256, 256, 0, stream>>>(tanhW, 128,  8, frTanh);
  pack_frags_kernel<<<(2  * 4 * 256) / 256, 256, 0, stream>>>(linW,   24,  2, frLin);

  labels_kernel<<<(N_WIN * OUT_L + 255) / 256, 256, 0, stream>>>(x, levels, wsf, noise_la, out);

  lstm_head_kernel<<<N_WGS, 256, 0, stream>>>(
      x, noise_in, levels, wsf, frWh0, frWx1, frWh1, frTanh, frLin,
      Wx0, b0, b1, tanhb, linb, out);
}
